// HybridModel_14379550507693
// MI455X (gfx1250) — compile-verified, matrix operand order checked by reading, not run on hardware
//
#include <hip/hip_runtime.h>
#include <hip/hip_bf16.h>
#include <math.h>

// ---------------- WMMA plumbing (gfx1250, wave32, bf16 16x16x32) ------------

typedef __attribute__((ext_vector_type(16))) __bf16 v16bf;
typedef __attribute__((ext_vector_type(8)))  float  v8f;

union FragU { unsigned short us[16]; uint4 q[2]; v16bf v; };

__device__ __forceinline__ unsigned short f2bf_bits(float f) {
  union { float f; unsigned u; } a; a.f = f;
  unsigned r = a.u + 0x7FFFu + ((a.u >> 16) & 1u);   // round-to-nearest-even
  return (unsigned short)(r >> 16);
}

__device__ __forceinline__ v8f wmma_bf16(v16bf a, v16bf b, v8f c) {
  return __builtin_amdgcn_wmma_f32_16x16x32_bf16(
      /*neg_a=*/false, a, /*neg_b=*/false, b,
      /*c_mod=*/(short)0, c, /*reuse_a=*/false, /*reuse_b=*/false);
}

// K-contiguous bf16 fragment: element (r,k) at P[r*sR + k].
// 16-bit 16x32 frag: lanes 0-15 -> rows, elems 0..7 = K{0..7}, 8..15 = K{16..23};
// lanes 16-31 -> K{8..15} and K{24..31}.  Each half = 8 contiguous bf16 = 16B.
__device__ __forceinline__ v16bf frag_ld(const unsigned short* P,
                                         int r0, int sR, int k0) {
  int lane = threadIdx.x & 31;
  const unsigned short* base = P + (r0 + (lane & 15)) * sR
                                 + k0 + ((lane & 16) ? 8 : 0);
  FragU u;
  u.q[0] = *(const uint4*)(base);
  u.q[1] = *(const uint4*)(base + 16);
  return u.v;
}

// f32 source (LDS state matrix), K-contiguous, converted to bf16 on the fly.
__device__ __forceinline__ v16bf frag_ld_f32(const float* P,
                                             int r0, int sR, int k0) {
  int lane = threadIdx.x & 31;
  const float* base = P + (r0 + (lane & 15)) * sR + k0 + ((lane & 16) ? 8 : 0);
  float4 f0 = *(const float4*)(base);
  float4 f1 = *(const float4*)(base + 4);
  float4 f2 = *(const float4*)(base + 16);
  float4 f3 = *(const float4*)(base + 20);
  FragU u;
  u.us[0]  = f2bf_bits(f0.x); u.us[1]  = f2bf_bits(f0.y);
  u.us[2]  = f2bf_bits(f0.z); u.us[3]  = f2bf_bits(f0.w);
  u.us[4]  = f2bf_bits(f1.x); u.us[5]  = f2bf_bits(f1.y);
  u.us[6]  = f2bf_bits(f1.z); u.us[7]  = f2bf_bits(f1.w);
  u.us[8]  = f2bf_bits(f2.x); u.us[9]  = f2bf_bits(f2.y);
  u.us[10] = f2bf_bits(f2.z); u.us[11] = f2bf_bits(f2.w);
  u.us[12] = f2bf_bits(f3.x); u.us[13] = f2bf_bits(f3.y);
  u.us[14] = f2bf_bits(f3.z); u.us[15] = f2bf_bits(f3.w);
  return u.v;
}

// ---------------- Async global->LDS (gfx1250), with safe fallback ------------

#if defined(__has_builtin)
#if __has_builtin(__builtin_amdgcn_global_load_async_to_lds_b128) && \
    __has_builtin(__builtin_amdgcn_s_wait_asynccnt)
#define USE_ASYNC_LDS 1
#endif
#endif
#ifndef USE_ASYNC_LDS
#define USE_ASYNC_LDS 0
#endif

#if USE_ASYNC_LDS
typedef int vint4_t __attribute__((vector_size(16)));
typedef __attribute__((address_space(1))) vint4_t* gas_v4;
typedef __attribute__((address_space(3))) vint4_t* las_v4;
__device__ __forceinline__ void cp16(const float* g, float* l) {
  __builtin_amdgcn_global_load_async_to_lds_b128((gas_v4)g, (las_v4)l, 0, 0);
}
__device__ __forceinline__ void cp_wait() {
  __builtin_amdgcn_s_wait_asynccnt(0);
}
#else
__device__ __forceinline__ void cp16(const float* g, float* l) {
  *(float4*)l = *(const float4*)g;
}
__device__ __forceinline__ void cp_wait() {}
#endif

// ---------------- Model constants -------------------------------------------

#define MDL_B  8
#define MDL_T  4096
#define MDL_M  (MDL_B * MDL_T)   // 32768 token rows
#define MDL_D  256
#define MDL_H  4
#define MDL_HK 32
#define MDL_HV 64
#define MDL_CH 64
#define MDL_NCH (MDL_T / MDL_CH)

#define EPI_BIAS    1
#define EPI_GELU    2
#define EPI_LOGSIG  4
#define EPI_RESID   8
#define EPI_BF16OUT 16

// ---------------- Small kernels ---------------------------------------------

__global__ void embed_kernel(const int* __restrict__ tok,
                             const float* __restrict__ emb,
                             float* __restrict__ X) {
  long m = blockIdx.x; int d = threadIdx.x;
  X[m * MDL_D + d] = emb[(long)tok[m] * MDL_D + d];
}

// Weight transpose + bf16 convert + zero pad:  W[K,N] f32 -> Wt[Npad,Kpad] bf16
__global__ void wt_kernel(const float* __restrict__ W, unsigned short* __restrict__ Wt,
                          int K, int N, int Kpad, int Npad) {
  int idx = blockIdx.x * 256 + threadIdx.x;
  if (idx >= Npad * Kpad) return;
  int n = idx / Kpad, k = idx % Kpad;
  Wt[idx] = (n < N && k < K) ? f2bf_bits(W[(long)k * N + n]) : (unsigned short)0;
}

__global__ void conv_bf16_kernel(const float* __restrict__ src,
                                 unsigned short* __restrict__ dst) {
  long i = (long)blockIdx.x * 256 + threadIdx.x;
  dst[i] = f2bf_bits(src[i]);
}

// LayerNorm, bf16 output (feeds WMMA GEMMs)
__global__ void ln_kernel(const float* __restrict__ X,
                          const float* __restrict__ w,
                          const float* __restrict__ bvec,
                          unsigned short* __restrict__ Ho) {
  int row = blockIdx.x, tid = threadIdx.x;   // 256 threads, D=256
  __shared__ float red[256];
  float v = X[(long)row * MDL_D + tid];
  red[tid] = v; __syncthreads();
  for (int s = 128; s > 0; s >>= 1) { if (tid < s) red[tid] += red[tid + s]; __syncthreads(); }
  float mu = red[0] * (1.0f / MDL_D);
  __syncthreads();
  float d = v - mu;
  red[tid] = d * d; __syncthreads();
  for (int s = 128; s > 0; s >>= 1) { if (tid < s) red[tid] += red[tid + s]; __syncthreads(); }
  float var = red[0] * (1.0f / MDL_D);
  Ho[(long)row * MDL_D + tid] = f2bf_bits(d * rsqrtf(var + 1e-5f) * w[tid] + bvec[tid]);
}

// ---------------- WMMA GEMM --------------------------------------------------
// C[M,N] = epi(A[M,K]bf16 @ Wt[N,K]bf16^T); one wave = 32x32 output tile
// (2 m-frags x 2 n-frags -> 4 WMMA per 8 b128 loads per K-step).

template<int EPI>
__device__ __forceinline__ void gemm_store(v8f acc, int mt, int nt, int N,
                                           const float* bias, const float* R,
                                           float* Cf, unsigned short* Cb, int lane) {
  int cn = nt + (lane & 15);
  int mb = mt + ((lane & 16) ? 8 : 0);
#pragma unroll
  for (int e = 0; e < 8; ++e) {
    float v = acc[e];
    if (EPI & EPI_BIAS)   v += bias[cn];
    if (EPI & EPI_GELU)   v = 0.5f * v * (1.0f + erff(v * 0.70710678118f));
    if (EPI & EPI_LOGSIG) v = (fminf(v, 0.0f) - log1pf(expf(-fabsf(v)))) * (1.0f / 16.0f);
    long idx = (long)(mb + e) * N + cn;
    if (EPI & EPI_RESID)  v += R[idx];
    if (EPI & EPI_BF16OUT) Cb[idx] = f2bf_bits(v);
    else                   Cf[idx] = v;
  }
}

template<int EPI>
__global__ __launch_bounds__(128)
void gemm_bf16_kernel(const unsigned short* __restrict__ A,
                      const unsigned short* __restrict__ Wt,
                      const float* __restrict__ bias, const float* __restrict__ R,
                      float* __restrict__ Cf, unsigned short* __restrict__ Cb,
                      int M, int N, int Kd) {
  int wave = threadIdx.x >> 5;
  int lane = threadIdx.x & 31;
  int m0 = (blockIdx.y * 4 + wave) * 32;
  int n0 = blockIdx.x * 32;
  v8f a00 = {}, a01 = {}, a10 = {}, a11 = {};
  for (int k0 = 0; k0 < Kd; k0 += 32) {
    __builtin_prefetch(A  + (m0 + (lane & 15)) * Kd + k0 + 256, 0, 1);
    __builtin_prefetch(Wt + (n0 + (lane & 15)) * Kd + k0 + 256, 0, 1);
    v16bf fa0 = frag_ld(A,  m0,      Kd, k0);
    v16bf fa1 = frag_ld(A,  m0 + 16, Kd, k0);
    v16bf fb0 = frag_ld(Wt, n0,      Kd, k0);
    v16bf fb1 = frag_ld(Wt, n0 + 16, Kd, k0);
    a00 = wmma_bf16(fa0, fb0, a00);
    a01 = wmma_bf16(fa0, fb1, a01);
    a10 = wmma_bf16(fa1, fb0, a10);
    a11 = wmma_bf16(fa1, fb1, a11);
  }
  gemm_store<EPI>(a00, m0,      n0,      N, bias, R, Cf, Cb, lane);
  gemm_store<EPI>(a01, m0,      n0 + 16, N, bias, R, Cf, Cb, lane);
  gemm_store<EPI>(a10, m0 + 16, n0,      N, bias, R, Cf, Cb, lane);
  gemm_store<EPI>(a11, m0 + 16, n0 + 16, N, bias, R, Cf, Cb, lane);
}

// ---------------- GLA chunked attention --------------------------------------
// One workgroup per (b,h); 8 waves; recurrent state ST[dv=64][dk=32] (f32, LDS);
// 64 chunks of 64 tokens.  Next chunk's raw q/k/v/gk streamed into a
// double-buffered LDS staging area with async global->LDS copies, overlapped
// with the current chunk's WMMA work.

// Copy one chunk (64 tokens) of q/k/gk [64x32 f32] and v [64x64 f32] to LDS.
__device__ __forceinline__ void copy_chunk(int tid,
    const float* gq, const float* gk, const float* gg, const float* gv,
    float* dq, float* dk, float* dg, float* dv) {
  for (int u = tid; u < 512; u += 256) {      // 64 rows x 8 x 16B
    int t = u >> 3, s = (u & 7) * 4;
    cp16(gq + (long)t * (MDL_H * MDL_HK) + s, dq + t * 32 + s);
    cp16(gk + (long)t * (MDL_H * MDL_HK) + s, dk + t * 32 + s);
    cp16(gg + (long)t * (MDL_H * MDL_HK) + s, dg + t * 32 + s);
  }
  for (int u = tid; u < 1024; u += 256) {     // 64 rows x 16 x 16B
    int t = u >> 4, s = (u & 15) * 4;
    cp16(gv + (long)t * (MDL_H * MDL_HV) + s, dv + t * 64 + s);
  }
}

__global__ __launch_bounds__(256)
void gla_kernel(const float* __restrict__ Qg, const float* __restrict__ Kg,
                const float* __restrict__ Vg, const float* __restrict__ GKg,
                float* __restrict__ Og) {
  const int bh = blockIdx.x, b = bh >> 2, h = bh & 3;
  const int tid = threadIdx.x, wave = tid >> 5, lane = tid & 31;
  const float scale = 0.17677669529663687f;            // HK^-0.5

  __shared__ __align__(16) float ST[64 * 32];          // state^T [dv][dk]
  __shared__ __align__(16) float bc[64 * 32];          // cum-log-decay [t][dk]
  __shared__ float glast[32];
  __shared__ __align__(16) unsigned short qs[64 * 32];   // [t][dk]
  __shared__ __align__(16) unsigned short ki[64 * 32];   // [t][dk]
  __shared__ __align__(16) unsigned short keT[32 * 64];  // [dk][t]
  __shared__ __align__(16) unsigned short vtT[64 * 64];  // [dv][t]
  __shared__ __align__(16) unsigned short attb[64 * 64]; // [tq][tk]
  // raw-f32 double-buffered staging for async copies
  __shared__ __align__(16) float sQ[2][64 * 32];
  __shared__ __align__(16) float sK[2][64 * 32];
  __shared__ __align__(16) float sG[2][64 * 32];
  __shared__ __align__(16) float sV[2][64 * 64];

  const long baseQ = ((long)b * MDL_T * MDL_H + h) * MDL_HK;  // + t*H*HK
  const long baseV = ((long)b * MDL_T * MDL_H + h) * MDL_HV;  // + t*H*HV

  for (int i = tid; i < 64 * 32; i += 256) ST[i] = 0.0f;

  // prologue: stage chunk 0
  copy_chunk(tid, Qg + baseQ, Kg + baseQ, GKg + baseQ, Vg + baseV,
             sQ[0], sK[0], sG[0], sV[0]);

  for (int nch = 0; nch < MDL_NCH; ++nch) {
    const int cur = nch & 1;
    cp_wait();            // own async copies done
    __syncthreads();      // -> everyone's copies visible; prev readers done

    // kick off next chunk's staging (overlaps with all compute below)
    if (nch + 1 < MDL_NCH) {
      long t1 = (long)(nch + 1) * MDL_CH;
      copy_chunk(tid, Qg + baseQ + t1 * (MDL_H * MDL_HK),
                      Kg + baseQ + t1 * (MDL_H * MDL_HK),
                      GKg + baseQ + t1 * (MDL_H * MDL_HK),
                      Vg + baseV + t1 * (MDL_H * MDL_HV),
                 sQ[1 - cur], sK[1 - cur], sG[1 - cur], sV[1 - cur]);
    }

    // ---- stage 1: gate cumsum + decayed q/k/v (all reads from LDS staging) --
    const float* cQ = sQ[cur]; const float* cK = sK[cur];
    const float* cG = sG[cur]; const float* cV = sV[cur];
    if (tid < 32) {
      int d = tid; float acc = 0.0f;
      for (int t = 0; t < 64; ++t) { acc += cG[t * 32 + d]; bc[t * 32 + d] = acc; }
      glast[d] = expf(acc);
    }
    __syncthreads();
    for (int i = tid; i < 64 * 32; i += 256) {
      int t = i >> 5, d = i & 31;
      float bv = bc[i];
      float qv = cQ[i], kvv = cK[i];
      qs[i] = f2bf_bits(qv * expf(bv) * scale);
      ki[i] = f2bf_bits(kvv * expf(-bv));
      keT[d * 64 + t] = f2bf_bits(kvv * expf(bc[63 * 32 + d] - bv));
    }
    for (int i = tid; i < 64 * 64; i += 256) {
      int t = i >> 6, dv = i & 63;
      vtT[dv * 64 + t] = f2bf_bits(cV[i]);
    }
    __syncthreads();

    // ---- stage 2: att = tril(q_s @ k_i^T)  [64x64, K=32] ----
    for (int tt = wave; tt < 16; tt += 8) {
      int mi = tt >> 2, ni = tt & 3;
      v8f c = {};
      c = wmma_bf16(frag_ld(qs, mi * 16, 32, 0),
                    frag_ld(ki, ni * 16, 32, 0), c);
      int cn = ni * 16 + (lane & 15);
      int mb = mi * 16 + ((lane & 16) ? 8 : 0);
#pragma unroll
      for (int e = 0; e < 8; ++e) {
        int m = mb + e;
        attb[m * 64 + cn] = (cn <= m) ? f2bf_bits(c[e]) : (unsigned short)0;
      }
    }
    __syncthreads();

    // ---- stage 3: o = att @ v + q_s @ S  [64x64] ----
    int t0 = nch * MDL_CH;
    for (int tt = wave; tt < 16; tt += 8) {
      int mi = tt >> 2, ni = tt & 3;
      v8f c = {};
      c = wmma_bf16(frag_ld(attb, mi * 16, 64,  0),
                    frag_ld(vtT,  ni * 16, 64,  0), c);
      c = wmma_bf16(frag_ld(attb, mi * 16, 64, 32),
                    frag_ld(vtT,  ni * 16, 64, 32), c);
      c = wmma_bf16(frag_ld(qs,   mi * 16, 32,  0),
                    frag_ld_f32(ST, ni * 16, 32, 0), c);
      int cn = ni * 16 + (lane & 15);
      int mb = mi * 16 + ((lane & 16) ? 8 : 0);
#pragma unroll
      for (int e = 0; e < 8; ++e) {
        int m = mb + e;
        Og[(((long)b * MDL_T + t0 + m) * MDL_H + h) * MDL_HV + cn] = c[e];
      }
    }
    __syncthreads();

    // ---- stage 4: S = S*diag(g_last) + k_e^T @ v  [32x64, K=64] ----
    {
      int mi = wave >> 2, ni = wave & 3;   // 8 waves -> 8 tiles (m=dk, n=dv)
      v8f c = {};
      c = wmma_bf16(frag_ld(keT, mi * 16, 64,  0),
                    frag_ld(vtT, ni * 16, 64,  0), c);
      c = wmma_bf16(frag_ld(keT, mi * 16, 64, 32),
                    frag_ld(vtT, ni * 16, 64, 32), c);
      int cn = ni * 16 + (lane & 15);
      int mb = mi * 16 + ((lane & 16) ? 8 : 0);
#pragma unroll
      for (int e = 0; e < 8; ++e) {
        int m = mb + e;
        ST[cn * 32 + m] = ST[cn * 32 + m] * glast[m] + c[e];
      }
    }
    __syncthreads();
  }
}

// Fused RMSNorm + swish gate -> bf16 (feeds wo GEMM)
__global__ void gate_kernel(const float* __restrict__ O, const float* __restrict__ G,
                            const float* __restrict__ gnw,
                            unsigned short* __restrict__ Ob) {
  long row = blockIdx.x; int tid = threadIdx.x;   // 64 threads per (b,t,h)
  __shared__ float red[64];
  float o = O[row * 64 + tid];
  red[tid] = o * o; __syncthreads();
  for (int s = 32; s > 0; s >>= 1) { if (tid < s) red[tid] += red[tid + s]; __syncthreads(); }
  float rms = rsqrtf(red[0] * (1.0f / 64.0f) + 1e-5f);
  float g = G[row * 64 + tid];
  float sig = 1.0f / (1.0f + expf(-g));
  Ob[row * 64 + tid] = f2bf_bits(o * rms * gnw[tid] * g * sig);
}

// ---------------- Host orchestration ----------------------------------------

extern "C" void kernel_launch(void* const* d_in, const int* in_sizes, int n_in,
                              void* d_out, int out_size, void* d_ws, size_t ws_size,
                              hipStream_t stream) {
  const int*   tokens = (const int*)  d_in[0];
  const float* embedw = (const float*)d_in[1];
  const float* ln_w   = (const float*)d_in[2];
  const float* ln_b   = (const float*)d_in[3];
  const float* wq     = (const float*)d_in[4];
  const float* wk     = (const float*)d_in[5];
  const float* wv     = (const float*)d_in[6];
  const float* wg     = (const float*)d_in[7];
  const float* gk_w1  = (const float*)d_in[8];
  const float* gk_w2  = (const float*)d_in[9];
  const float* gk_b2  = (const float*)d_in[10];
  const float* gnorm  = (const float*)d_in[11];
  const float* wo     = (const float*)d_in[12];
  const float* mlp_w1 = (const float*)d_in[13];
  const float* mlp_b1 = (const float*)d_in[14];
  const float* mlp_w2 = (const float*)d_in[15];
  const float* mlp_b2 = (const float*)d_in[16];
  const float* head_w = (const float*)d_in[17];
  const float* head_b = (const float*)d_in[18];
  (void)in_sizes; (void)n_in; (void)out_size; (void)ws_size;

  const int M = MDL_M;
  typedef unsigned short us_t;
  float* X     = (float*)d_ws;                 // [M,256] f32 residual stream
  us_t*  Hb    = (us_t*)(X + (size_t)M * 256); // [M,256] bf16
  float* Qb    = (float*)(Hb + (size_t)M * 256);
  float* Kb    = Qb + (size_t)M * 128;
  float* Vb    = Kb + (size_t)M * 128;
  float* Gb    = Vb + (size_t)M * 256;
  us_t*  GK16B = (us_t*)(Gb + (size_t)M * 256);   // [M,32] bf16 (padded)
  float* GKb   = (float*)(GK16B + (size_t)M * 32);
  float* Ob    = GKb + (size_t)M * 128;           // [M,256] f32 attention out
  us_t*  ObB   = (us_t*)(Ob + (size_t)M * 256);   // [M,256] bf16 gated out
  us_t*  MID   = (us_t*)Qb;                       // alias: [M,512] bf16 (q/k dead)
  us_t*  XB    = ObB;                             // alias for head input
  us_t*  wtb   = ObB + (size_t)M * 256;           // transposed-weight region

  // Transposed bf16 weight buffers (Npad x Kpad each)
  auto wt = [&](const float* W, us_t* dst, int K, int N, int Kpad, int Npad) {
    int tot = Npad * Kpad;
    wt_kernel<<<(tot + 255) / 256, 256, 0, stream>>>(W, dst, K, N, Kpad, Npad);
  };
  us_t* wqT[2];  us_t* wkT[2];  us_t* wvT[2];  us_t* wgT[2];
  us_t* g1T[2];  us_t* g2T[2];  us_t* woT[2];  us_t* w1T[2];  us_t* w2T[2];
  us_t* p = wtb;
  for (int i = 0; i < 2; ++i) {
    wqT[i] = p; p += 128 * 256;
    wkT[i] = p; p += 128 * 256;
    wvT[i] = p; p += 256 * 256;
    wgT[i] = p; p += 256 * 256;
    g1T[i] = p; p += 32 * 256;
    g2T[i] = p; p += 128 * 32;
    woT[i] = p; p += 256 * 256;
    w1T[i] = p; p += 512 * 256;
    w2T[i] = p; p += 256 * 512;
  }
  us_t* headT = p;
  for (int i = 0; i < 2; ++i) {
    wt(wq     + (size_t)i * 256 * 128, wqT[i], 256, 128, 256, 128);
    wt(wk     + (size_t)i * 256 * 128, wkT[i], 256, 128, 256, 128);
    wt(wv     + (size_t)i * 256 * 256, wvT[i], 256, 256, 256, 256);
    wt(wg     + (size_t)i * 256 * 256, wgT[i], 256, 256, 256, 256);
    wt(gk_w1  + (size_t)i * 256 * 16,  g1T[i], 256, 16,  256, 32);
    wt(gk_w2  + (size_t)i * 16 * 128,  g2T[i], 16,  128, 32,  128);
    wt(wo     + (size_t)i * 256 * 256, woT[i], 256, 256, 256, 256);
    wt(mlp_w1 + (size_t)i * 256 * 512, w1T[i], 256, 512, 256, 512);
    wt(mlp_w2 + (size_t)i * 512 * 256, w2T[i], 512, 256, 512, 256);
  }
  wt(head_w, headT, 256, 128, 256, 128);

  auto gemm = [&](const us_t* A, const us_t* Wt, const float* bias,
                  const float* R, float* Cf, us_t* Cb,
                  int Mm, int Nn, int Kk, int epi) {
    dim3 grid(Nn / 32, Mm / 128);
    dim3 blk(128);
#define LG(E) gemm_bf16_kernel<E><<<grid, blk, 0, stream>>>(A, Wt, bias, R, Cf, Cb, Mm, Nn, Kk)
    switch (epi) {
      case 0:                                   LG(0); break;
      case EPI_BF16OUT:                         LG(EPI_BF16OUT); break;
      case EPI_BIAS | EPI_LOGSIG:               LG(EPI_BIAS | EPI_LOGSIG); break;
      case EPI_RESID:                           LG(EPI_RESID); break;
      case EPI_BIAS | EPI_GELU | EPI_BF16OUT:   LG(EPI_BIAS | EPI_GELU | EPI_BF16OUT); break;
      case EPI_BIAS | EPI_RESID:                LG(EPI_BIAS | EPI_RESID); break;
      case EPI_BIAS:                            LG(EPI_BIAS); break;
    }
#undef LG
  };

  embed_kernel<<<M, 256, 0, stream>>>(tokens, embedw, X);

  for (int i = 0; i < 2; ++i) {
    const float* gb2_i = gk_b2  + (size_t)i * 128;
    const float* gn_i  = gnorm  + (size_t)i * 64;
    const float* b1_i  = mlp_b1 + (size_t)i * 512;
    const float* b2_i  = mlp_b2 + (size_t)i * 256;
    const float* lw_i  = ln_w   + (size_t)i * 256;
    const float* lb_i  = ln_b   + (size_t)i * 256;

    // attention block
    ln_kernel<<<M, 256, 0, stream>>>(X, lw_i, lb_i, Hb);
    gemm(Hb, wqT[i], nullptr, nullptr, Qb, nullptr, M, 128, 256, 0);
    gemm(Hb, wkT[i], nullptr, nullptr, Kb, nullptr, M, 128, 256, 0);
    gemm(Hb, wvT[i], nullptr, nullptr, Vb, nullptr, M, 256, 256, 0);
    gemm(Hb, wgT[i], nullptr, nullptr, Gb, nullptr, M, 256, 256, 0);
    gemm(Hb, g1T[i], nullptr, nullptr, nullptr, GK16B, M, 32, 256, EPI_BF16OUT);
    gemm(GK16B, g2T[i], gb2_i, nullptr, GKb, nullptr, M, 128, 32,
         EPI_BIAS | EPI_LOGSIG);
    gla_kernel<<<MDL_B * MDL_H, 256, 0, stream>>>(Qb, Kb, Vb, GKb, Ob);
    gate_kernel<<<(long)M * MDL_H, 64, 0, stream>>>(Ob, Gb, gn_i, ObB);
    gemm(ObB, woT[i], nullptr, X, X, nullptr, M, 256, 256, EPI_RESID);

    // MLP block
    ln_kernel<<<M, 256, 0, stream>>>(X, lw_i, lb_i, Hb);
    gemm(Hb, w1T[i], b1_i, nullptr, nullptr, MID, M, 512, 256,
         EPI_BIAS | EPI_GELU | EPI_BF16OUT);
    gemm(MID, w2T[i], b2_i, X, X, nullptr, M, 256, 512, EPI_BIAS | EPI_RESID);
  }

  conv_bf16_kernel<<<M, 256, 0, stream>>>(X, XB);
  gemm(XB, headT, head_b, nullptr, (float*)d_out, nullptr, M, 128, 256, EPI_BIAS);
}
